// BiLSTM_45612552684169
// MI455X (gfx1250) — matrix operand order repaired and checked by reading, not compiled, not run on hardware
//
#include <hip/hip_runtime.h>
#include <hip/hip_bf16.h>
#include <math.h>

// ---------------------------------------------------------------------------
// Types for CDNA5 WMMA (gfx1250, wave32)
// ---------------------------------------------------------------------------
typedef __attribute__((ext_vector_type(16))) __bf16 v16bf;
typedef __attribute__((ext_vector_type(8)))  __bf16 v8bf;
typedef __attribute__((ext_vector_type(8)))  float  v8f;

// f32 -> bf16 with round-to-nearest-even (bit-exact, no reliance on cast codegen)
__device__ __forceinline__ __bf16 f2bf(float f) {
    union { float f; unsigned u; } x; x.f = f;
    unsigned r = x.u + 0x7FFFu + ((x.u >> 16) & 1u);
    unsigned short h = (unsigned short)(r >> 16);
    __bf16 out;
    __builtin_memcpy(&out, &h, 2);
    return out;
}

__device__ __forceinline__ float sigmoidf_(float x) { return 1.0f / (1.0f + __expf(-x)); }

// ---------------------------------------------------------------------------
// 1) Transpose Whh [2,1000,250] -> [2,250,1000] so the scan GEMV is coalesced
// ---------------------------------------------------------------------------
__global__ void k_transpose_whh(float* __restrict__ dst, const float* __restrict__ src,
                                int G /*1000*/, int H /*250*/) {
    int idx = blockIdx.x * blockDim.x + threadIdx.x;
    int total = 2 * G * H;
    if (idx >= total) return;
    int k = idx % H;
    int j = (idx / H) % G;
    int d = idx / (H * G);
    dst[((size_t)d * H + k) * G + j] = src[idx]; // src[d][j][k]
}

// ---------------------------------------------------------------------------
// 2) Layer-0 input projection: xz0[seq][t][d*1000+j] = b0 + x[t] . Wih0[d][j]
// ---------------------------------------------------------------------------
__global__ void k_proj0(const float* __restrict__ vr, const float* __restrict__ vl,
                        const float* __restrict__ wih0, const float* __restrict__ b0,
                        float* __restrict__ xz, int T /*768*/, int D /*20*/) {
    const int G2 = 2000;
    int idx = blockIdx.x * blockDim.x + threadIdx.x;
    int total = 2 * T * G2;
    if (idx >= total) return;
    int n   = idx % G2;
    int t   = (idx / G2) % T;
    int seq = idx / (G2 * T);
    const float* x = (seq ? vl : vr) + (size_t)t * D;
    const float* w = wih0 + (size_t)n * D;   // [2,1000,20] flat: row n, n = d*1000+j
    float acc = b0[n];                       // b0 flat [2000]
    #pragma unroll
    for (int k = 0; k < 20; ++k) acc = fmaf(x[k], w[k], acc);
    xz[idx] = acc;
}

// ---------------------------------------------------------------------------
// 3) LSTM scan. One workgroup per (seq, dir). h, z shared in LDS; c private.
//    xz laid out [seq][t][2000] (dir-major halves); hs out [seq][t][500].
// ---------------------------------------------------------------------------
__global__ __launch_bounds__(1024) void k_lstm_scan(
    const float* __restrict__ xz, const float* __restrict__ whhT,
    float* __restrict__ hs, int T /*768*/) {
    const int H = 250, G = 1000;
    int seq = blockIdx.x >> 1;
    int d   = blockIdx.x & 1;
    const float* xzb = xz   + (size_t)seq * T * 2 * G + (size_t)d * G;
    const float* wT  = whhT + (size_t)d * H * G;     // [250][1000]
    float*       hsb = hs   + (size_t)seq * T * 2 * H + (size_t)d * H;

    __shared__ float h_lds[256];
    __shared__ float z_lds[1024];
    int j = threadIdx.x;
    if (j < 256) h_lds[j] = 0.0f;
    float c = 0.0f;
    __syncthreads();

    for (int s = 0; s < T; ++s) {
        int t = d ? (T - 1 - s) : s;       // bwd dir walks reversed time
        if (j < G) {
            float z = xzb[(size_t)t * 2 * G + j];
            const float* wcol = wT + j;
            #pragma unroll 5
            for (int k = 0; k < H; ++k)
                z = fmaf(wcol[(size_t)k * G], h_lds[k], z);
            z_lds[j] = z;
        }
        __syncthreads();
        if (j < H) {
            float zi = z_lds[j], zf = z_lds[j + H], zg = z_lds[j + 2 * H], zo = z_lds[j + 3 * H];
            float ig = sigmoidf_(zi);
            float fg = sigmoidf_(zf);
            float g  = tanhf(zg);
            float og = sigmoidf_(zo);
            c = fmaf(fg, c, ig * g);
            float h = og * tanhf(c);
            h_lds[j] = h;
            hsb[(size_t)t * 2 * H + j] = h;
        }
        __syncthreads();
    }
}

// ---------------------------------------------------------------------------
// 4) f32 -> bf16 conversion with K padding (zeros) and optional column offset
// ---------------------------------------------------------------------------
__global__ void k_cvt_pad(__bf16* __restrict__ dst, const float* __restrict__ src,
                          int rows, int dcols, int ccols, int sstride, int soff) {
    int idx = blockIdx.x * blockDim.x + threadIdx.x;
    long total = (long)rows * dcols;
    if (idx >= total) return;
    int cc = idx % dcols;
    int r  = idx / dcols;
    float v = (cc < ccols) ? src[(size_t)r * sstride + soff + cc] : 0.0f;
    dst[idx] = f2bf(v);
}

// ---------------------------------------------------------------------------
// 5) bf16 WMMA GEMM: C[M,N] = act(A[M,K] @ B[N,K]^T + bias).
//    K multiple of 32; M,N multiples of 16. One 16x16 C tile per wave.
//    Fragment layouts used here (wave32; contiguous k order for A/B):
//      A: lane L -> row m=L%16; elems e: k = e + 16*(L/16)
//      B: lane L -> col n=L%16; elems e: k = e + 16*(L/16)
//      C: lane L, reg r -> (m = r + 8*(L/16), n = L%16)
// ---------------------------------------------------------------------------
__global__ __launch_bounds__(128) void k_gemm_bf16_wmma(
    float* __restrict__ C, const __bf16* __restrict__ A, const __bf16* __restrict__ B,
    const float* __restrict__ bias, int M, int N, int K, int relu) {
    int wave = threadIdx.x >> 5;
    int lane = threadIdx.x & 31;
    int ntn  = N >> 4;
    int tile = blockIdx.x * 4 + wave;
    if (tile >= (M >> 4) * ntn) return;           // uniform per wave: EXEC stays all-ones
    int tm = tile / ntn, tn = tile % ntn;
    int hi = lane >> 4, lo = lane & 15;

    const __bf16* arow = A + (size_t)(tm * 16 + lo) * K;
    const __bf16* brow = B + (size_t)(tn * 16 + lo) * K + 16 * hi;

    v8f acc = {};
    for (int kb = 0; kb < K; kb += 32) {
        __builtin_prefetch(brow + kb + 128, 0, 1);     // global_prefetch_b8
        union { v16bf v; v8bf h[2]; } a;
        a.h[0] = *(const v8bf*)(arow + kb + 16 * hi);
        a.h[1] = *(const v8bf*)(arow + kb + 16 * hi + 8);
        v16bf b = *(const v16bf*)(brow + kb);
        acc = __builtin_amdgcn_wmma_f32_16x16x32_bf16(
            /*neg_a=*/false, a.v, /*neg_b=*/false, b,
            /*c_mod=*/(short)0, acc, /*reuse_a=*/false, /*reuse_b=*/false);
    }

    int n  = tn * 16 + lo;
    int m0 = tm * 16 + 8 * hi;
    float bv = bias ? bias[n] : 0.0f;
    #pragma unroll
    for (int r = 0; r < 8; ++r) {
        float v = acc[r] + bv;
        if (relu) v = v > 0.0f ? v : 0.0f;
        C[(size_t)(m0 + r) * N + n] = v;
    }
}

// ---------------------------------------------------------------------------
// 6) Fused pairwise head: out[i,j,:] = log_softmax(relu(pr[i]+pl[j]+b3) @ Wo^T + bo)
//    16x16 pair tiles; pr/pl/b3/Wo staged in LDS (pad 257 kills bank conflicts).
// ---------------------------------------------------------------------------
__global__ __launch_bounds__(256) void k_pairwise(
    const float* __restrict__ pr, const float* __restrict__ pl,
    const float* __restrict__ b3, const float* __restrict__ Wo,
    const float* __restrict__ bo, float* __restrict__ out, int NL /*768*/) {
    const int C = 256;
    __shared__ float spr[16][257];
    __shared__ float spl[16][257];
    __shared__ float sb3[256], sw0[256], sw1[256];

    int t  = threadIdx.y * 16 + threadIdx.x;   // 0..255
    int i0 = blockIdx.x * 16, j0 = blockIdx.y * 16;
    #pragma unroll 4
    for (int r = 0; r < 16; ++r) {
        spr[r][t] = pr[(size_t)(i0 + r) * C + t];
        spl[r][t] = pl[(size_t)(j0 + r) * C + t];
    }
    sb3[t] = b3[t];
    sw0[t] = Wo[t];
    sw1[t] = Wo[C + t];
    __syncthreads();

    int ti = threadIdx.y, tj = threadIdx.x;
    float a0 = 0.0f, a1 = 0.0f;
    #pragma unroll 8
    for (int c = 0; c < C; ++c) {
        float v = spr[ti][c] + spl[tj][c] + sb3[c];
        v = v > 0.0f ? v : 0.0f;
        a0 = fmaf(v, sw0[c], a0);
        a1 = fmaf(v, sw1[c], a1);
    }
    a0 += bo[0]; a1 += bo[1];
    float mx  = fmaxf(a0, a1);
    float lse = mx + logf(__expf(a0 - mx) + __expf(a1 - mx));
    size_t o = ((size_t)(i0 + ti) * NL + (j0 + tj)) * 2;
    out[o]     = a0 - lse;
    out[o + 1] = a1 - lse;
}

// ---------------------------------------------------------------------------
// Host orchestration
// ---------------------------------------------------------------------------
static inline int ceil_div(long a, int b) { return (int)((a + b - 1) / b); }

extern "C" void kernel_launch(void* const* d_in, const int* in_sizes, int n_in,
                              void* d_out, int out_size, void* d_ws, size_t ws_size,
                              hipStream_t stream) {
    const float* v_r   = (const float*)d_in[0];
    const float* v_l   = (const float*)d_in[1];
    const float* w_ih0 = (const float*)d_in[2];
    const float* w_hh0 = (const float*)d_in[3];
    const float* b0    = (const float*)d_in[4];
    const float* w_ih1 = (const float*)d_in[5];
    const float* w_hh1 = (const float*)d_in[6];
    const float* b1    = (const float*)d_in[7];
    const float* W1    = (const float*)d_in[8];
    const float* b_1   = (const float*)d_in[9];
    const float* W2    = (const float*)d_in[10];
    const float* b_2   = (const float*)d_in[11];
    const float* W3    = (const float*)d_in[12];
    const float* b_3   = (const float*)d_in[13];
    const float* Wo    = (const float*)d_in[14];
    const float* b_o   = (const float*)d_in[15];
    float* out = (float*)d_out;

    const int T = 768, D = 20, H = 250, G = 1000;
    const int M = 2 * T;               // 1536 rows (v_r then v_l)

    // --- scratch carve-out (256B aligned); ~45 MB total ---
    char* base = (char*)d_ws;
    size_t off = 0;
    auto alloc = [&](size_t bytes) -> char* {
        char* p = base + off;
        off = (off + bytes + 255) & ~(size_t)255;
        return p;
    };
    float*  xz      = (float*) alloc((size_t)2 * T * 2000 * 4); // reused for layer0 and layer1
    float*  hs0     = (float*) alloc((size_t)M * 500 * 4);
    float*  hs1     = (float*) alloc((size_t)M * 500 * 4);
    float*  whhT0   = (float*) alloc((size_t)2 * H * G * 4);
    float*  whhT1   = (float*) alloc((size_t)2 * H * G * 4);
    __bf16* hs0_bf  = (__bf16*)alloc((size_t)M * 512 * 2);
    __bf16* wih1_bf = (__bf16*)alloc((size_t)2000 * 512 * 2);
    __bf16* hs1_bf  = (__bf16*)alloc((size_t)M * 512 * 2);
    __bf16* w1_bf   = (__bf16*)alloc((size_t)1024 * 512 * 2);
    float*  h1      = (float*) alloc((size_t)M * 1024 * 4);
    __bf16* h1_bf   = (__bf16*)alloc((size_t)M * 1024 * 2);
    __bf16* w2_bf   = (__bf16*)alloc((size_t)512 * 1024 * 2);
    float*  h2      = (float*) alloc((size_t)M * 512 * 4);
    __bf16* h2_bf   = (__bf16*)alloc((size_t)M * 512 * 2);
    __bf16* w3a_bf  = (__bf16*)alloc((size_t)256 * 512 * 2);
    __bf16* w3b_bf  = (__bf16*)alloc((size_t)256 * 512 * 2);
    float*  pr      = (float*) alloc((size_t)T * 256 * 4);
    float*  pl      = (float*) alloc((size_t)T * 256 * 4);
    (void)ws_size; (void)in_sizes; (void)n_in; (void)out_size;

    const int TPB = 256;
    auto gemm = [&](float* Cm, const __bf16* Am, const __bf16* Bm, const float* bias,
                    int Mm, int Nm, int Km, int relu) {
        int tiles = (Mm / 16) * (Nm / 16);
        k_gemm_bf16_wmma<<<ceil_div(tiles, 4), 128, 0, stream>>>(Cm, Am, Bm, bias, Mm, Nm, Km, relu);
    };

    // Pre-transpose recurrent weights for coalesced GEMV in the scan
    k_transpose_whh<<<ceil_div((long)2 * G * H, TPB), TPB, 0, stream>>>(whhT0, w_hh0, G, H);
    k_transpose_whh<<<ceil_div((long)2 * G * H, TPB), TPB, 0, stream>>>(whhT1, w_hh1, G, H);

    // Layer 0: input projection + bidirectional scan
    k_proj0<<<ceil_div((long)2 * T * 2000, TPB), TPB, 0, stream>>>(v_r, v_l, w_ih0, b0, xz, T, D);
    k_lstm_scan<<<4, 1024, 0, stream>>>(xz, whhT0, hs0, T);

    // Layer 1: WMMA input projection (xz = hs0 @ Wih1^T + b1, both dirs fused N=2000)
    k_cvt_pad<<<ceil_div((long)M * 512, TPB), TPB, 0, stream>>>(hs0_bf, hs0, M, 512, 500, 500, 0);
    k_cvt_pad<<<ceil_div((long)2000 * 512, TPB), TPB, 0, stream>>>(wih1_bf, w_ih1, 2000, 512, 500, 500, 0);
    gemm(xz, hs0_bf, wih1_bf, b1, M, 2000, 512, 0);
    k_lstm_scan<<<4, 1024, 0, stream>>>(xz, whhT1, hs1, T);

    // MLP: h1 = relu(hs1 @ W1^T + b_1), h2 = relu(h1 @ W2^T + b_2)   [WMMA]
    k_cvt_pad<<<ceil_div((long)M * 512, TPB), TPB, 0, stream>>>(hs1_bf, hs1, M, 512, 500, 500, 0);
    k_cvt_pad<<<ceil_div((long)1024 * 512, TPB), TPB, 0, stream>>>(w1_bf, W1, 1024, 512, 500, 500, 0);
    gemm(h1, hs1_bf, w1_bf, b_1, M, 1024, 512, 1);

    k_cvt_pad<<<ceil_div((long)M * 1024, TPB), TPB, 0, stream>>>(h1_bf, h1, M, 1024, 1024, 1024, 0);
    k_cvt_pad<<<ceil_div((long)512 * 1024, TPB), TPB, 0, stream>>>(w2_bf, W2, 512, 1024, 1024, 1024, 0);
    gemm(h2, h1_bf, w2_bf, b_2, M, 512, 1024, 1);

    // Split projections: pr = r2 @ W3[:, :512]^T ; pl = l2 @ W3[:, 512:]^T   [WMMA]
    k_cvt_pad<<<ceil_div((long)M * 512, TPB), TPB, 0, stream>>>(h2_bf, h2, M, 512, 512, 512, 0);
    k_cvt_pad<<<ceil_div((long)256 * 512, TPB), TPB, 0, stream>>>(w3a_bf, W3, 256, 512, 512, 1024, 0);
    k_cvt_pad<<<ceil_div((long)256 * 512, TPB), TPB, 0, stream>>>(w3b_bf, W3, 256, 512, 512, 1024, 512);
    gemm(pr, h2_bf,                     w3a_bf, nullptr, T, 256, 512, 0);
    gemm(pl, h2_bf + (size_t)T * 512,   w3b_bf, nullptr, T, 256, 512, 0);

    // Fused pairwise relu + Wo + log_softmax
    dim3 pb(16, 16), pg(T / 16, T / 16);
    k_pairwise<<<pg, pb, 0, stream>>>(pr, pl, b_3, Wo, b_o, out, T);
}
